// OffloadedLinear4bit_45535243272235
// MI455X (gfx1250) — compile-verified, matrix operand order checked by reading, not run on hardware
//
#include <hip/hip_runtime.h>

typedef __attribute__((ext_vector_type(16))) __bf16   v16bf;
typedef __attribute__((ext_vector_type(8)))  float    v8f;
typedef __attribute__((ext_vector_type(4)))  float    f4v;
typedef __attribute__((ext_vector_type(4)))  int      i4v;
typedef __attribute__((ext_vector_type(4)))  unsigned u4v;

#define IN_F   4096
#define OUT_F  11008
#define M_TOT  8192          // 4 * 2048
#define KS     64            // staged K per loop iteration (2 WMMA substeps)
#define MT     128           // block tile M
#define NT     128           // block tile N
#define NIT    (IN_F / KS)   // 64 staging steps
#define RSTRD  (KS / 2)      // LDS row stride in dwords (32)

__constant__ float NF4_CB[16] = {
    -1.0f, -0.6961928009986877f, -0.5250730514526367f, -0.39491748809814453f,
    -0.28444138169288635f, -0.18477343022823334f, -0.09105003625154495f, 0.0f,
    0.07958029955625534f, 0.16093020141124725f, 0.24611230194568634f,
    0.33791524171829224f, 0.44070982933044434f, 0.5626170039176941f,
    0.7229568362236023f, 1.0f};

// round-to-nearest-even fp32 -> bf16, pack two into a dword
__device__ __forceinline__ unsigned pack2_bf16(float a, float b) {
    unsigned ua = __float_as_uint(a);
    unsigned ub = __float_as_uint(b);
    ua = (ua + 0x7fffu + ((ua >> 16) & 1u)) >> 16;
    ub = (ub + 0x7fffu + ((ub >> 16) & 1u)) >> 16;
    return (ua & 0xffffu) | (ub << 16);
}

union FragU { v16bf v; u4v q[2]; };

__launch_bounds__(256)
__global__ void nf4_wmma_gemm(const float* __restrict__ x,
                              const int*   __restrict__ codes,
                              const float* __restrict__ absmax,
                              float*       __restrict__ out) {
    // ping-pong 128x64 bf16 tiles, row-major, packed dword pairs (32 dwords/row)
    __shared__ unsigned lA[2][MT * RSTRD];   // x tiles   (2 x 16 KB)
    __shared__ unsigned lB[2][NT * RSTRD];   // W tiles   (2 x 16 KB)
    __shared__ unsigned lutp[256];           // NF4 pair codebook, packed bf16 (1 KB)

    const int tid  = threadIdx.x;
    const int lane = tid & 31;
    const int wave = tid >> 5;

    // pair codebook: lutp[c0 | c1<<4] = bf16(cb[c0]) | bf16(cb[c1])<<16
    lutp[tid] = pack2_bf16(NF4_CB[tid & 15], NF4_CB[tid >> 4]);

    // ---- cooperative loader coordinates: thread -> (row 0..127, k-chunk 0/32)
    const int lrow = tid >> 1;
    const int lkc  = (tid & 1) << 5;          // 0 or 32 (one NF4 half-block each)

    const long mrow = (long)blockIdx.y * MT + lrow;   // row of x
    const long nrow = (long)blockIdx.x * NT + lrow;   // row of W (out feature)
    const float* xptr = x     + mrow * IN_F + lkc;
    const int*   cptr = codes + nrow * IN_F + lkc;
    const int    bblk = (int)nrow * (IN_F / 64);      // absmax block base for this W row
    const unsigned soff = (unsigned)lrow * RSTRD + (lkc >> 1);

    // ---- compute coordinates: wave -> 64x32 sub-tile; lane -> WMMA layout
    const int waveM = (wave >> 2) * 64;   // 0 or 64
    const int waveN = (wave & 3) * 32;    // 0..96
    const int fr    = lane & 15;          // row within a 16-tile
    const int fq    = (lane >> 4) * 4;    // dword offset: k-halves {0-7} or {8-15}

    v8f acc[4][2] = {};

    // prefetch registers (overwritten each iteration after consumption)
    f4v gx0, gx1, gx2, gx3, gx4, gx5, gx6, gx7;
    i4v gc0, gc1, gc2, gc3, gc4, gc5, gc6, gc7;
    float gs;

    auto load_tile = [&](int k0) {
        gx0 = *(const f4v*)(xptr + k0);       gx1 = *(const f4v*)(xptr + k0 + 4);
        gx2 = *(const f4v*)(xptr + k0 + 8);   gx3 = *(const f4v*)(xptr + k0 + 12);
        gx4 = *(const f4v*)(xptr + k0 + 16);  gx5 = *(const f4v*)(xptr + k0 + 20);
        gx6 = *(const f4v*)(xptr + k0 + 24);  gx7 = *(const f4v*)(xptr + k0 + 28);
        gc0 = *(const i4v*)(cptr + k0);       gc1 = *(const i4v*)(cptr + k0 + 4);
        gc2 = *(const i4v*)(cptr + k0 + 8);   gc3 = *(const i4v*)(cptr + k0 + 12);
        gc4 = *(const i4v*)(cptr + k0 + 16);  gc5 = *(const i4v*)(cptr + k0 + 20);
        gc6 = *(const i4v*)(cptr + k0 + 24);  gc7 = *(const i4v*)(cptr + k0 + 28);
        gs  = absmax[bblk + ((k0 + lkc) >> 6)];
    };

    auto pk4 = [](f4v a, f4v b) -> u4v {
        u4v r;
        r.x = pack2_bf16(a.x, a.y);  r.y = pack2_bf16(a.z, a.w);
        r.z = pack2_bf16(b.x, b.y);  r.w = pack2_bf16(b.z, b.w);
        return r;
    };

    auto pack_store = [&](int p) {
        // x tile: fp32 -> bf16 (manual RNE pack)
        *(u4v*)&lA[p][soff]      = pk4(gx0, gx1);
        *(u4v*)&lA[p][soff + 4]  = pk4(gx2, gx3);
        *(u4v*)&lA[p][soff + 8]  = pk4(gx4, gx5);
        *(u4v*)&lA[p][soff + 12] = pk4(gx6, gx7);
        // W tile: NF4 dequant entirely in bf16: LUT pair * splatted absmax
        const unsigned gs2 = pack2_bf16(gs, gs);
        auto dq1 = [&](int ca, int cb_) -> unsigned {
            unsigned e = lutp[(ca & 15) | ((cb_ & 15) << 4)];
            unsigned r;
            asm("v_pk_mul_bf16 %0, %1, %2" : "=v"(r) : "v"(e), "v"(gs2));
            return r;
        };
        auto dqq = [&](i4v cA, i4v cB) -> u4v {
            u4v r;
            r.x = dq1(cA.x, cA.y);  r.y = dq1(cA.z, cA.w);
            r.z = dq1(cB.x, cB.y);  r.w = dq1(cB.z, cB.w);
            return r;
        };
        *(u4v*)&lB[p][soff]      = dqq(gc0, gc1);
        *(u4v*)&lB[p][soff + 4]  = dqq(gc2, gc3);
        *(u4v*)&lB[p][soff + 8]  = dqq(gc4, gc5);
        *(u4v*)&lB[p][soff + 12] = dqq(gc6, gc7);
    };

    // ---- prologue: stage tile 0 into buffer 0
    load_tile(0);
    __syncthreads();          // lutp visible
    pack_store(0);

    for (int it = 0; it < NIT; ++it) {
        const int p = it & 1;

        // issue next tile's global loads (wrap on last iter; in flight during WMMAs)
        load_tile(((it + 1) & (NIT - 1)) * KS);

        __syncthreads();      // RAW: buf p staged by previous iteration / prologue

        // ---------- compute: 2 K-substeps x 8 WMMAs from buf p ----------
#pragma unroll
        for (int s = 0; s < 2; ++s) {
            const unsigned cb = (unsigned)s * 16 + fq;
            FragU fb0, fb1, fa;
            const unsigned b0 = (unsigned)(waveN + 0  + fr) * RSTRD + cb;
            const unsigned b1 = (unsigned)(waveN + 16 + fr) * RSTRD + cb;
            fb0.q[0] = *(u4v*)&lB[p][b0];  fb0.q[1] = *(u4v*)&lB[p][b0 + 8];
            fb1.q[0] = *(u4v*)&lB[p][b1];  fb1.q[1] = *(u4v*)&lB[p][b1 + 8];
#pragma unroll
            for (int t = 0; t < 4; ++t) {
                const unsigned a0 = (unsigned)(waveM + t * 16 + fr) * RSTRD + cb;
                fa.q[0] = *(u4v*)&lA[p][a0];  fa.q[1] = *(u4v*)&lA[p][a0 + 8];
                acc[t][0] = __builtin_amdgcn_wmma_f32_16x16x32_bf16(
                    false, fa.v, false, fb0.v, (short)0, acc[t][0], false, false);
                acc[t][1] = __builtin_amdgcn_wmma_f32_16x16x32_bf16(
                    false, fa.v, false, fb1.v, (short)0, acc[t][1], false, false);
            }
        }

        // ---------- pack the in-flight tile, stage into buf 1-p ----------
        // s_wait_loadcnt lands here, after the WMMA block -> latency hidden.
        pack_store(p ^ 1);
    }

    // ---------- store: C/D layout -> row-major out [8192, 11008] ----------
    const long onb = (long)blockIdx.x * NT + waveN + fr;
    const long omb = (long)blockIdx.y * MT + waveM + ((lane >> 4) << 3);
#pragma unroll
    for (int t = 0; t < 4; ++t) {
#pragma unroll
        for (int u = 0; u < 2; ++u) {
            const long n  = onb + u * 16;
            const long mb = omb + t * 16;
#pragma unroll
            for (int i = 0; i < 8; ++i)
                out[(mb + i) * (long)OUT_F + n] = acc[t][u][i];
        }
    }
}

extern "C" void kernel_launch(void* const* d_in, const int* in_sizes, int n_in,
                              void* d_out, int out_size, void* d_ws, size_t ws_size,
                              hipStream_t stream) {
    (void)in_sizes; (void)n_in; (void)out_size; (void)d_ws; (void)ws_size;
    const float* x      = (const float*)d_in[0];
    const int*   codes  = (const int*)d_in[1];
    const float* absmax = (const float*)d_in[2];
    float*       out    = (float*)d_out;

    dim3 grid(OUT_F / NT, M_TOT / MT);   // 86 x 64 blocks
    nf4_wmma_gemm<<<grid, 256, 0, stream>>>(x, codes, absmax, out);
}
